// CapsNet_60756607369946
// MI455X (gfx1250) — matrix operand order repaired and checked by reading, not compiled, not run on hardware
//
#include <hip/hip_runtime.h>
#include <hip/hip_bf16.h>
#include <math.h>

typedef __attribute__((ext_vector_type(16))) _Float16 v16h;
typedef __attribute__((ext_vector_type(8)))  float    v8f;

#define B_SZ 512
#define NUM_ROUTES 1152
#define NUM_CAPS 10

__device__ __forceinline__ v8f wmma16x16x32(v16h a, v16h b, v8f c) {
  return __builtin_amdgcn_wmma_f32_16x16x32_f16(
      /*neg_a=*/false, a, /*neg_b=*/false, b,
      /*c_mod=*/(short)0, c, /*reuse_a=*/false, /*reuse_b=*/false);
}

// ---------------- utility kernels ----------------

__global__ void k_f2h(const float* __restrict__ in, _Float16* __restrict__ out, int n) {
  int i = blockIdx.x * blockDim.x + threadIdx.x;
  if (i < n) out[i] = (_Float16)in[i];
}

__global__ void k_zero(float* __restrict__ p, int n) {
  int i = blockIdx.x * blockDim.x + threadIdx.x;
  if (i < n) p[i] = 0.0f;
}

__global__ void k_copy(const float* __restrict__ in, float* __restrict__ out, int n) {
  int i = blockIdx.x * blockDim.x + threadIdx.x;
  if (i < n) out[i] = in[i];
}

// ---------------- conv1 prep: padded weights + padded f16 im2col ----------------
// Wp[m][k] : 256 x 96 (k >= 81 zero)
__global__ void k_padw1(const float* __restrict__ W, _Float16* __restrict__ Wp) {
  int idx = blockIdx.x * blockDim.x + threadIdx.x;   // 256*96
  if (idx >= 256 * 96) return;
  int k = idx % 96, m = idx / 96;
  Wp[idx] = (k < 81) ? (_Float16)W[m * 81 + k] : (_Float16)0.0f;
}

// pat[b][p][k] : 512 x 400 x 96 (transposed im2col so fragment loads are contiguous)
__global__ void k_im2col1(const float* __restrict__ data, _Float16* __restrict__ pat) {
  int idx = blockIdx.x * blockDim.x + threadIdx.x;   // 512*400*96
  if (idx >= 512 * 400 * 96) return;
  int k = idx % 96; int rest = idx / 96;
  int p = rest % 400; int b = rest / 400;
  _Float16 v = (_Float16)0.0f;
  if (k < 81) {
    int oy = p / 20, ox = p % 20, ky = k / 9, kx = k % 9;
    v = (_Float16)data[(size_t)b * 784 + (oy + ky) * 28 + (ox + kx)];
  }
  pat[idx] = v;
}

// ---------------- conv1 GEMM: C[256 x 400] = Wp[256 x 96] * pat^T, relu, f16 ----------------
__global__ void k_conv1g(const _Float16* __restrict__ Wp, const float* __restrict__ bias,
                         const _Float16* __restrict__ pat, _Float16* __restrict__ out) {
  const int ntile = blockIdx.x;   // 0..24
  const int mtile = blockIdx.y;   // 0..15
  const int b     = blockIdx.z;   // 0..511
  const int lane  = threadIdx.x;  // 0..31
  const int hs = lane >> 4;
  const int lm = lane & 15;

  const _Float16* arow = Wp + (size_t)(mtile * 16 + lm) * 96;
  const _Float16* brow = pat + ((size_t)b * 400 + ntile * 16 + lm) * 96;
  const int p = ntile * 16 + lm;

  v8f acc = {0.f,0.f,0.f,0.f,0.f,0.f,0.f,0.f};
  #pragma unroll
  for (int ks = 0; ks < 96; ks += 32) {
    v16h af, bf;
    const _Float16* ap = arow + ks + hs * 16;
    const _Float16* bp = brow + ks + hs * 16;
    #pragma unroll
    for (int i = 0; i < 16; ++i) af[i] = ap[i];
    #pragma unroll
    for (int i = 0; i < 16; ++i) bf[i] = bp[i];
    acc = wmma16x16x32(af, bf, acc);
  }
  #pragma unroll
  for (int j = 0; j < 8; ++j) {
    int ch = mtile * 16 + j + hs * 8;
    float r = fmaxf(acc[j] + bias[ch], 0.0f);
    out[((size_t)b * 256 + ch) * 400 + p] = (_Float16)r;
  }
}

// ---------------- conv2 (primary caps): WMMA + LDS-staged B, 16 waves/block -------
// grid (3 ntiles, 512 images), block 512 = 16 waves; wave w computes mtile w.
// B tile (32K x 16N) is identical across the 16 waves -> staged once into LDS.
__global__ __launch_bounds__(512) void k_conv2(const _Float16* __restrict__ Wh,
                                               const float* __restrict__ bias,
                                               const _Float16* __restrict__ x1,
                                               float* __restrict__ u) {
  const int ntile = blockIdx.x;   // 0..2
  const int b     = blockIdx.y;   // 0..511
  const int t     = threadIdx.x;  // 0..511
  const int wave  = t >> 5;       // mtile
  const int lane  = t & 31;
  const int hs = lane >> 4;
  const int lm = lane & 15;

  const _Float16* img = x1 + (size_t)b * 256 * 400;

  // staging coords: thread t fills LDS slot t = (frag lane)*16 + (frag elem)
  const int sl = t >> 4;                  // fragment lane 0..31
  const int si = t & 15;                  // fragment element 0..15
  const int sn = sl & 15;                 // column within N-tile
  const int sk = ((sl >> 4) << 4) + si;   // local K 0..31
  const int sp = ntile * 16 + sn;         // output pixel
  const bool pv = (sp < 36);
  const int base2 = pv ? ((2 * (sp / 6)) * 20 + 2 * (sp % 6)) : 0;

  auto gather = [&](int ks) -> _Float16 {
    if (!pv) return (_Float16)0.0f;
    unsigned k  = (unsigned)(ks * 32 + sk);
    unsigned c  = k / 81u;
    unsigned rm = k - 81u * c;
    unsigned ky = rm / 9u;
    unsigned kx = rm - 9u * ky;
    return img[(size_t)c * 400 + base2 + ky * 20 + kx];
  };

  __shared__ __align__(32) _Float16 bs[2][512];

  const _Float16* wrow = Wh + (size_t)(wave * 16 + lm) * 20736;

  bs[0][t] = gather(0);
  __syncthreads();

  v8f acc = {0.f,0.f,0.f,0.f,0.f,0.f,0.f,0.f};
  for (int ks = 0; ks < 648; ++ks) {      // K = 20736 = 648*32
    const int cur = ks & 1;
    if (ks + 1 < 648) bs[cur ^ 1][t] = gather(ks + 1);

    v16h bf = *((const v16h*)&bs[cur][lane * 16]);   // ds_load_b128 x2
    v16h af;
    const _Float16* ap = wrow + ks * 32 + hs * 16;
    #pragma unroll
    for (int i = 0; i < 16; ++i) af[i] = ap[i];
    __builtin_prefetch(ap + 8 * 32, 0, 1);           // next K-slices of A

    acc = wmma16x16x32(af, bf, acc);
    __syncthreads();
  }

  const int p = ntile * 16 + lm;
  if (p < 36) {
    #pragma unroll
    for (int j = 0; j < 8; ++j) {
      int ch = wave * 16 + j + hs * 8;
      // flat (ch*36 + p) == (route, i) flattening of reshape(8,32,6,6)->(1152,8)
      u[(size_t)b * 9216 + (size_t)ch * 36 + p] = acc[j] + bias[ch];
    }
  }
}

// ---------------- squash over groups of 8 (in place) ----------------
__global__ void k_squash8(float* __restrict__ u, int n) {
  int i = blockIdx.x * blockDim.x + threadIdx.x;
  if (i >= n) return;
  float* p = u + (size_t)i * 8;
  float sq = 0.0f;
  #pragma unroll
  for (int j = 0; j < 8; ++j) sq += p[j] * p[j];
  float sc = sq / ((1.0f + sq) * sqrtf(sq));
  #pragma unroll
  for (int j = 0; j < 8; ++j) p[j] *= sc;
}

// ---------------- u_hat[b,r,c,o] = sum_i W[r,c,o,i] * u[b,r,i] ----------------
__global__ void k_uhat(const float* __restrict__ W, const float* __restrict__ u,
                       float* __restrict__ uhat) {
  int r = blockIdx.x, b = blockIdx.y, co = threadIdx.x;  // co in 0..159
  const float* uu = u + ((size_t)b * NUM_ROUTES + r) * 8;
  const float* ww = W + ((size_t)r * 160 + co) * 8;
  float a = 0.0f;
  #pragma unroll
  for (int i = 0; i < 8; ++i) a += ww[i] * uu[i];
  uhat[((size_t)b * NUM_ROUTES + r) * 160 + co] = a;
}

// ---------------- softmax over routes; bij/cij layout [b][c][r] ----------------
__global__ void k_softmax_routes(const float* __restrict__ bij, float* __restrict__ cij) {
  int g = blockIdx.x;          // b*10 + c
  int t = threadIdx.x;         // 256
  const float* x = bij + (size_t)g * NUM_ROUTES;
  float* y = cij + (size_t)g * NUM_ROUTES;
  __shared__ float red[256];
  float mx = -3.0e38f;
  for (int r = t; r < NUM_ROUTES; r += 256) mx = fmaxf(mx, x[r]);
  red[t] = mx; __syncthreads();
  for (int s = 128; s > 0; s >>= 1) { if (t < s) red[t] = fmaxf(red[t], red[t + s]); __syncthreads(); }
  mx = red[0]; __syncthreads();
  float sum = 0.0f;
  for (int r = t; r < NUM_ROUTES; r += 256) sum += expf(x[r] - mx);
  red[t] = sum; __syncthreads();
  for (int s = 128; s > 0; s >>= 1) { if (t < s) red[t] += red[t + s]; __syncthreads(); }
  float inv = 1.0f / red[0];
  for (int r = t; r < NUM_ROUTES; r += 256) y[r] = expf(x[r] - mx) * inv;
}

// ---------------- s_j[b,c,o] = sum_r c[b,c,r] * uhat[b,r,c,o] ----------------
__global__ void k_sj(const float* __restrict__ cij, const float* __restrict__ uhat,
                     float* __restrict__ sj) {
  int c = blockIdx.x, b = blockIdx.y;
  int o = threadIdx.x, part = threadIdx.y;     // 16 x 16
  const float* cc = cij + ((size_t)b * NUM_CAPS + c) * NUM_ROUTES;
  float acc = 0.0f;
  for (int r = part; r < NUM_ROUTES; r += 16)
    acc += cc[r] * uhat[(((size_t)b * NUM_ROUTES + r) * NUM_CAPS + c) * 16 + o];
  __shared__ float red[16][16];
  red[part][o] = acc; __syncthreads();
  if (part == 0) {
    float s = 0.0f;
    #pragma unroll
    for (int q = 0; q < 16; ++q) s += red[q][o];
    sj[((size_t)b * NUM_CAPS + c) * 16 + o] = s;
  }
}

// ---------------- global sum of squares over 81920 elems ----------------
__global__ void k_sumsq(const float* __restrict__ s, float* __restrict__ out) {
  __shared__ float red[1024];
  int t = threadIdx.x;
  float a = 0.0f;
  for (int i = t; i < B_SZ * NUM_CAPS * 16; i += 1024) { float v = s[i]; a += v * v; }
  red[t] = a; __syncthreads();
  for (int k = 512; k > 0; k >>= 1) { if (t < k) red[t] += red[t + k]; __syncthreads(); }
  if (t == 0) out[0] = red[0];
}

__global__ void k_squashv(const float* __restrict__ s, const float* __restrict__ ssq,
                          float* __restrict__ v, int n) {
  int i = blockIdx.x * blockDim.x + threadIdx.x;
  if (i >= n) return;
  float sq = ssq[0];
  float sc = sq / ((1.0f + sq) * sqrtf(sq));
  v[i] = sc * s[i];
}

// ---------------- b_ij[b,c,r] += sum_o uhat[b,r,c,o]*v[b,c,o] ----------------
__global__ void k_aij(const float* __restrict__ uhat, const float* __restrict__ v,
                      float* __restrict__ bij) {
  size_t idx = (size_t)blockIdx.x * blockDim.x + threadIdx.x;
  if (idx >= (size_t)B_SZ * NUM_CAPS * NUM_ROUTES) return;
  int r = (int)(idx % NUM_ROUTES);
  size_t bc = idx / NUM_ROUTES;
  int c = (int)(bc % NUM_CAPS);
  int b = (int)(bc / NUM_CAPS);
  const float* uh = uhat + (((size_t)b * NUM_ROUTES + r) * NUM_CAPS + c) * 16;
  const float* vv = v + ((size_t)b * NUM_CAPS + c) * 16;
  float a = 0.0f;
  #pragma unroll
  for (int o = 0; o < 16; ++o) a += uh[o] * vv[o];
  bij[idx] += a;
}

// ---------------- class norms, batch softmax, argmax/mask/t ----------------
__global__ void k_classnorm(const float* __restrict__ v, float* __restrict__ norm) {
  int i = blockIdx.x * blockDim.x + threadIdx.x;   // b*10+c
  if (i >= B_SZ * NUM_CAPS) return;
  const float* p = v + (size_t)i * 16;
  float s = 0.0f;
  #pragma unroll
  for (int o = 0; o < 16; ++o) s += p[o] * p[o];
  norm[i] = sqrtf(s);
}

__global__ void k_softmax_batch(const float* __restrict__ norm, float* __restrict__ sm) {
  int c = blockIdx.x;          // 0..9
  int t = threadIdx.x;         // 0..511 (= batch index)
  __shared__ float red[512];
  float x = norm[(size_t)t * NUM_CAPS + c];
  red[t] = x; __syncthreads();
  for (int s = 256; s > 0; s >>= 1) { if (t < s) red[t] = fmaxf(red[t], red[t + s]); __syncthreads(); }
  float mx = red[0]; __syncthreads();
  float e = expf(x - mx);
  red[t] = e; __syncthreads();
  for (int s = 256; s > 0; s >>= 1) { if (t < s) red[t] += red[t + s]; __syncthreads(); }
  sm[(size_t)t * NUM_CAPS + c] = e / red[0];
}

__global__ void k_argmask(const float* __restrict__ sm, const float* __restrict__ v,
                          float* __restrict__ out_mask, _Float16* __restrict__ th) {
  int b = blockIdx.x * blockDim.x + threadIdx.x;
  if (b >= B_SZ) return;
  int best = 0; float bv = sm[(size_t)b * NUM_CAPS];
  #pragma unroll
  for (int c = 1; c < NUM_CAPS; ++c) {
    float x = sm[(size_t)b * NUM_CAPS + c];
    if (x > bv) { bv = x; best = c; }
  }
  #pragma unroll
  for (int c = 0; c < NUM_CAPS; ++c) out_mask[(size_t)b * NUM_CAPS + c] = (c == best) ? 1.0f : 0.0f;
  for (int co = 0; co < 160; ++co) {
    int c = co >> 4;
    float val = (c == best) ? v[(size_t)b * 160 + co] : 0.0f;
    th[(size_t)b * 160 + co] = (_Float16)val;
  }
}

// ---------------- decoder GEMM: C = act(A[MxK] * W[NxK]^T + bias) ----------------
// mode 0: relu -> f16 outh ; mode 1: sigmoid -> f32 outf
__global__ void k_gemm(const _Float16* __restrict__ A, const _Float16* __restrict__ Wt,
                       const float* __restrict__ bias, int M, int N, int K,
                       _Float16* __restrict__ outh, float* __restrict__ outf, int mode) {
  const int ntile = blockIdx.x;
  const int mtile = blockIdx.y;
  const int lane  = threadIdx.x;
  const int hs = lane >> 4;
  const int lm = lane & 15;

  const _Float16* arow = A  + (size_t)(mtile * 16 + lm) * K;
  const _Float16* brow = Wt + (size_t)(ntile * 16 + lm) * K;  // B[k][n] = W[n][k]

  v8f acc = {0.f,0.f,0.f,0.f,0.f,0.f,0.f,0.f};
  for (int ks = 0; ks < K; ks += 32) {
    v16h af, bf;
    const _Float16* ap = arow + ks + hs * 16;
    const _Float16* bp = brow + ks + hs * 16;
    #pragma unroll
    for (int i = 0; i < 16; ++i) af[i] = ap[i];
    #pragma unroll
    for (int i = 0; i < 16; ++i) bf[i] = bp[i];
    __builtin_prefetch(ap + 4 * 32, 0, 1);
    __builtin_prefetch(bp + 4 * 32, 0, 1);
    acc = wmma16x16x32(af, bf, acc);
  }
  int n = ntile * 16 + lm;
  float bn = bias[n];
  #pragma unroll
  for (int j = 0; j < 8; ++j) {
    int m = mtile * 16 + j + hs * 8;
    float val = acc[j] + bn;
    if (mode == 0) {
      outh[(size_t)m * N + n] = (_Float16)fmaxf(val, 0.0f);
    } else {
      outf[(size_t)m * N + n] = 1.0f / (1.0f + expf(-val));
    }
  }
}

// ---------------- host launch ----------------

extern "C" void kernel_launch(void* const* d_in, const int* in_sizes, int n_in,
                              void* d_out, int out_size, void* d_ws, size_t ws_size,
                              hipStream_t stream) {
  const float* data   = (const float*)d_in[0];
  const float* conv_w = (const float*)d_in[1];
  const float* conv_b = (const float*)d_in[2];
  const float* pc_w   = (const float*)d_in[3];
  const float* pc_b   = (const float*)d_in[4];
  const float* W_dc   = (const float*)d_in[5];
  const float* w1 = (const float*)d_in[6];  const float* b1 = (const float*)d_in[7];
  const float* w2 = (const float*)d_in[8];  const float* b2 = (const float*)d_in[9];
  const float* w3 = (const float*)d_in[10]; const float* b3 = (const float*)d_in[11];

  float* out_v     = (float*)d_out;            // 512*10*16   = 81920
  float* out_recon = (float*)d_out + 81920;    // 512*784     = 401408
  float* out_mask  = (float*)d_out + 483328;   // 512*10      = 5120

  char* ws = (char*)d_ws;
  size_t off = 0;
  auto carve = [&](size_t bytes) { size_t o = off; off += (bytes + 255) & ~(size_t)255; return o; };

  _Float16* x1h  = (_Float16*)(ws + carve((size_t)512 * 256 * 400 * 2));   // conv1 out (f16)
  _Float16* pat  = (_Float16*)(ws + carve((size_t)512 * 400 * 96 * 2));    // conv1 im2col (padded)
  _Float16* wp1  = (_Float16*)(ws + carve((size_t)256 * 96 * 2));          // conv1 weights (padded)
  float*    u    = (float*)   (ws + carve((size_t)512 * 9216 * 4));        // primary caps
  float*    uhat = (float*)   (ws + carve((size_t)512 * 1152 * 160 * 4));  // u_hat
  float*    bij  = (float*)   (ws + carve((size_t)512 * 10 * 1152 * 4));
  float*    cij  = (float*)   (ws + carve((size_t)512 * 10 * 1152 * 4));
  float*    sj   = (float*)   (ws + carve((size_t)512 * 160 * 4));
  float*    v    = (float*)   (ws + carve((size_t)512 * 160 * 4));
  float*    ssq  = (float*)   (ws + carve(256));
  float*    nrm  = (float*)   (ws + carve((size_t)512 * 10 * 4));
  float*    sm   = (float*)   (ws + carve((size_t)512 * 10 * 4));
  _Float16* th   = (_Float16*)(ws + carve((size_t)512 * 160 * 2));
  _Float16* h1   = (_Float16*)(ws + carve((size_t)512 * 512 * 2));
  _Float16* h2   = (_Float16*)(ws + carve((size_t)512 * 1024 * 2));
  _Float16* pcwh = (_Float16*)(ws + carve((size_t)256 * 20736 * 2));
  _Float16* w1h  = (_Float16*)(ws + carve((size_t)512 * 160 * 2));
  _Float16* w2h  = (_Float16*)(ws + carve((size_t)1024 * 512 * 2));
  _Float16* w3h  = (_Float16*)(ws + carve((size_t)784 * 1024 * 2));
  (void)ws_size; (void)in_sizes; (void)n_in; (void)out_size;

  // conversions / prep
  k_f2h<<<(256 * 20736 + 255) / 256, 256, 0, stream>>>(pc_w, pcwh, 256 * 20736);
  k_f2h<<<(512 * 160 + 255) / 256, 256, 0, stream>>>(w1, w1h, 512 * 160);
  k_f2h<<<(1024 * 512 + 255) / 256, 256, 0, stream>>>(w2, w2h, 1024 * 512);
  k_f2h<<<(784 * 1024 + 255) / 256, 256, 0, stream>>>(w3, w3h, 784 * 1024);
  k_padw1<<<(256 * 96 + 255) / 256, 256, 0, stream>>>(conv_w, wp1);
  k_im2col1<<<(512 * 400 * 96 + 255) / 256, 256, 0, stream>>>(data, pat);
  k_zero<<<(512 * 10 * 1152 + 255) / 256, 256, 0, stream>>>(bij, 512 * 10 * 1152);

  // conv1 + relu (WMMA, fully contiguous fragments)
  k_conv1g<<<dim3(25, 16, 512), 32, 0, stream>>>(wp1, conv_b, pat, x1h);
  // primary caps conv (WMMA, LDS-staged B shared by 16 waves)
  k_conv2<<<dim3(3, 512), 512, 0, stream>>>(pcwh, pc_b, x1h, u);
  // squash per 8-vector (in place)
  k_squash8<<<(512 * 1152 + 255) / 256, 256, 0, stream>>>(u, 512 * 1152);
  // u_hat
  k_uhat<<<dim3(1152, 512), 160, 0, stream>>>(W_dc, u, uhat);

  // dynamic routing, 3 iterations
  for (int it = 0; it < 3; ++it) {
    k_softmax_routes<<<512 * 10, 256, 0, stream>>>(bij, cij);
    k_sj<<<dim3(10, 512), dim3(16, 16), 0, stream>>>(cij, uhat, sj);
    k_sumsq<<<1, 1024, 0, stream>>>(sj, ssq);
    k_squashv<<<(81920 + 255) / 256, 256, 0, stream>>>(sj, ssq, v, 81920);
    if (it < 2)
      k_aij<<<(512 * 10 * 1152 + 255) / 256, 256, 0, stream>>>(uhat, v, bij);
  }

  // output #1: v
  k_copy<<<(81920 + 255) / 256, 256, 0, stream>>>(v, out_v, 81920);

  // classes -> softmax over batch -> argmax mask (+ masked f16 t)
  k_classnorm<<<(512 * 10 + 255) / 256, 256, 0, stream>>>(v, nrm);
  k_softmax_batch<<<10, 512, 0, stream>>>(nrm, sm);
  k_argmask<<<2, 256, 0, stream>>>(sm, v, out_mask, th);

  // decoder MLP (WMMA GEMMs)
  k_gemm<<<dim3(512 / 16, 512 / 16), 32, 0, stream>>>(th, w1h, b1, 512, 512, 160, h1, nullptr, 0);
  k_gemm<<<dim3(1024 / 16, 512 / 16), 32, 0, stream>>>(h1, w2h, b2, 512, 1024, 512, h2, nullptr, 0);
  k_gemm<<<dim3(784 / 16, 512 / 16), 32, 0, stream>>>(h2, w3h, b3, 512, 784, 1024, nullptr, out_recon, 1);
}
